// SpeechTokenizer_77360950936169
// MI455X (gfx1250) — matrix operand order repaired and checked by reading, not compiled, not run on hardware
//
#include <hip/hip_runtime.h>

// ---------------------------------------------------------------------------
// VQ quantize (argmin over codebook) for MI455X / gfx1250.
//   dists[n,k] = ||x||^2 - 2 x.c + ||c||^2  ->  argmin_k == argmax_k (x.c - 0.5||c||^2)
// GEMM 32768x4096x512 via V_WMMA_F32_16X16X32_BF16 with bf16 hi/lo split
// (3 bf16 GEMMs ~= f32 accuracy), f32 accumulate. q gathered from the
// original f32 codebook => exact. Compute-bound: ~1.37e11 FLOP vs ~6us HBM.
// ---------------------------------------------------------------------------

typedef __attribute__((ext_vector_type(16))) __bf16 v16bf;
typedef __attribute__((ext_vector_type(8)))  __bf16 v8bf;
typedef __attribute__((ext_vector_type(8)))  float  v8f;

#define D_DIM      512
#define K_CODES    4096
#define ROWS_WG    128      // token rows per workgroup
#define NSUB       64       // codes per chunk (4 x 16-wide WMMA tiles)
#define LDS_STRIDE 520      // bf16 elems per LDS row: 512 + 8 pad (bank stagger)

__device__ __forceinline__ unsigned short f32_to_bf16(float f) {
    unsigned int u = __float_as_uint(f);
    u += 0x7fffu + ((u >> 16) & 1u);      // round-to-nearest-even
    return (unsigned short)(u >> 16);
}
__device__ __forceinline__ float bf16_to_f32(unsigned short h) {
    return __uint_as_float(((unsigned int)h) << 16);
}

// ---------------------------------------------------------------------------
// Pre-pass: codebook f32 [K,D] -> bf16 hi/lo [K,D] + halfnorm[k] = 0.5*||c||^2
// ---------------------------------------------------------------------------
__global__ void __launch_bounds__(256) vq_pack_codebook(
    const float* __restrict__ cb,
    unsigned short* __restrict__ hi,
    unsigned short* __restrict__ lo,
    float* __restrict__ halfnorm) {
    __shared__ float red[256];
    const int k = blockIdx.x, t = threadIdx.x;
    const float* row = cb + (size_t)k * D_DIM;
    float ss = 0.f;
    for (int c = t; c < D_DIM; c += 256) {
        float v = row[c];
        unsigned short hh = f32_to_bf16(v);
        unsigned short ll = f32_to_bf16(v - bf16_to_f32(hh));
        hi[(size_t)k * D_DIM + c] = hh;
        lo[(size_t)k * D_DIM + c] = ll;
        ss += v * v;
    }
    red[t] = ss;
    __syncthreads();
    for (int s = 128; s > 0; s >>= 1) {
        if (t < s) red[t] += red[t + s];
        __syncthreads();
    }
    if (t == 0) halfnorm[k] = 0.5f * red[0];
}

// ---------------------------------------------------------------------------
// Main kernel: 8 waves/WG, each wave owns 16 token rows, scans all 4096 codes.
// ---------------------------------------------------------------------------
union APack { v16bf v; v8bf h[2]; };

#define WMMA_BF16(A, B, C) \
    __builtin_amdgcn_wmma_f32_16x16x32_bf16(false, (A), false, (B), (short)0, (C), false, false)

__global__ void __launch_bounds__(256) vq_main(
    const float* __restrict__ hsrc,            // [N, 512] f32
    const float* __restrict__ cb,              // [4096, 512] f32 (for gather)
    const unsigned short* __restrict__ cbh,    // [4096, 512] bf16 hi
    const unsigned short* __restrict__ cbl,    // [4096, 512] bf16 lo
    const float* __restrict__ halfnorm,        // [4096]
    float* __restrict__ qout,                  // [N, 512] f32
    float* __restrict__ idxout) {              // [N] (stored as float)
    extern __shared__ char smem[];
    unsigned short* aH = (unsigned short*)smem;                  // 128*520 bf16
    unsigned short* aL = aH + ROWS_WG * LDS_STRIDE;              // 128*520 bf16
    int* bestLds = (int*)(aL + ROWS_WG * LDS_STRIDE);            // 128 ints

    const int tid = threadIdx.x;
    const int rowBase = blockIdx.x * ROWS_WG;

    // ---- Stage A tile into LDS as bf16 hi/lo (row-major, padded stride) ----
    {
        const float4* h4 = (const float4*)(hsrc + (size_t)rowBase * D_DIM);
        for (int i = tid; i < ROWS_WG * (D_DIM / 4); i += 256) {
            const int row = i >> 7;          // D_DIM/4 == 128
            const int c4  = i & 127;
            float4 v = h4[(size_t)row * (D_DIM / 4) + c4];
            unsigned short h0 = f32_to_bf16(v.x), h1 = f32_to_bf16(v.y);
            unsigned short h2 = f32_to_bf16(v.z), h3 = f32_to_bf16(v.w);
            unsigned short l0 = f32_to_bf16(v.x - bf16_to_f32(h0));
            unsigned short l1 = f32_to_bf16(v.y - bf16_to_f32(h1));
            unsigned short l2 = f32_to_bf16(v.z - bf16_to_f32(h2));
            unsigned short l3 = f32_to_bf16(v.w - bf16_to_f32(h3));
            const int base = row * LDS_STRIDE + c4 * 4;           // even => 4B aligned
            unsigned int* ph = (unsigned int*)(aH + base);
            ph[0] = (unsigned)h0 | ((unsigned)h1 << 16);
            ph[1] = (unsigned)h2 | ((unsigned)h3 << 16);
            unsigned int* pl = (unsigned int*)(aL + base);
            pl[0] = (unsigned)l0 | ((unsigned)l1 << 16);
            pl[1] = (unsigned)l2 | ((unsigned)l3 << 16);
        }
    }
    __syncthreads();

    const int lane = tid & 31;
    const int wave = tid >> 5;
    const int ln   = lane & 15;   // N index within 16x16 tile / A row select
    const int hb   = lane >> 4;   // K-half select (ISA 7.12.2 bf16 layouts)
    const int row0 = wave * 16;   // this wave's 16 rows within the WG tile

    // A fragment base: lane holds row (row0+ln); elems 0..7 at d0+hb*8,
    // elems 8..15 at d0+16+hb*8 (16-bit A 16x32 layout).
    const unsigned short* aRowH = aH + (row0 + ln) * LDS_STRIDE + hb * 8;
    const unsigned short* aRowL = aL + (row0 + ln) * LDS_STRIDE + hb * 8;

    float best[8];
    int   bestc[8];
#pragma unroll
    for (int r = 0; r < 8; ++r) { best[r] = -3.4e38f; bestc[r] = 0; }

    for (int chunk = 0; chunk < K_CODES / NSUB; ++chunk) {
        const int code0 = chunk * NSUB;
        v8f acc[4] = {};
        // B fragment base per 16-code group: lane holds column (code0+g*16+ln),
        // elems 0..7 at K hb*8, elems 8..15 at K 16+hb*8 (matches A layout).
        const unsigned short* pBH[4];
        const unsigned short* pBL[4];
#pragma unroll
        for (int g = 0; g < 4; ++g) {
            const size_t off = (size_t)(code0 + g * 16 + ln) * D_DIM + hb * 8;
            pBH[g] = cbh + off;
            pBL[g] = cbl + off;
        }
#pragma unroll 2
        for (int dstep = 0; dstep < D_DIM / 32; ++dstep) {
            const int d0 = dstep * 32;
            APack ahp, alp;
            ahp.h[0] = *(const v8bf*)(aRowH + d0);
            ahp.h[1] = *(const v8bf*)(aRowH + d0 + 16);
            alp.h[0] = *(const v8bf*)(aRowL + d0);
            alp.h[1] = *(const v8bf*)(aRowL + d0 + 16);
            const v16bf ah = ahp.v;
            const v16bf al = alp.v;
#pragma unroll
            for (int g = 0; g < 4; ++g) {
                APack bhp, blp;
                bhp.h[0] = *(const v8bf*)(pBH[g] + d0);
                bhp.h[1] = *(const v8bf*)(pBH[g] + d0 + 16);
                blp.h[0] = *(const v8bf*)(pBL[g] + d0);
                blp.h[1] = *(const v8bf*)(pBL[g] + d0 + 16);
                const v16bf bh = bhp.v;
                const v16bf bl = blp.v;
                acc[g] = WMMA_BF16(ah, bh, acc[g]);   // hi*hi
                acc[g] = WMMA_BF16(al, bh, acc[g]);   // lo*hi
                acc[g] = WMMA_BF16(ah, bl, acc[g]);   // hi*lo
            }
        }
        // ---- running argmax: score = x.c - 0.5||c||^2 ----
        float hn[4];
#pragma unroll
        for (int g = 0; g < 4; ++g) hn[g] = halfnorm[code0 + g * 16 + ln];
#pragma unroll
        for (int g = 0; g < 4; ++g) {
            const int code = code0 + g * 16 + ln;
#pragma unroll
            for (int r = 0; r < 8; ++r) {
                const float s = acc[g][r] - hn[g];
                if (s > best[r]) { best[r] = s; bestc[r] = code; }  // strict > keeps lowest idx
            }
        }
    }

    // ---- cross-lane merge over the 16 column residues (keeps hb fixed) ----
#pragma unroll
    for (int m = 1; m <= 8; m <<= 1) {
#pragma unroll
        for (int r = 0; r < 8; ++r) {
            const float ov = __shfl_xor(best[r], m, 32);
            const int   oc = __shfl_xor(bestc[r], m, 32);
            if (ov > best[r] || (ov == best[r] && oc < bestc[r])) {
                best[r] = ov; bestc[r] = oc;
            }
        }
    }
    if (ln == 0) {
#pragma unroll
        for (int r = 0; r < 8; ++r) bestLds[row0 + hb * 8 + r] = bestc[r];
    }
    __syncthreads();

    // ---- emit idx (as float) and gather exact f32 codebook rows ----
    if (tid < ROWS_WG) idxout[rowBase + tid] = (float)bestLds[tid];
    const float4* cb4 = (const float4*)cb;
    float4* q4 = (float4*)qout;
    for (int i = tid; i < ROWS_WG * (D_DIM / 4); i += 256) {
        const int row = i >> 7;
        const int c4  = i & 127;
        const int code = bestLds[row];
        q4[(size_t)(rowBase + row) * (D_DIM / 4) + c4] =
            cb4[(size_t)code * (D_DIM / 4) + c4];
    }
}

// ---------------------------------------------------------------------------
extern "C" void kernel_launch(void* const* d_in, const int* in_sizes, int n_in,
                              void* d_out, int out_size, void* d_ws, size_t ws_size,
                              hipStream_t stream) {
    const float* hsrc = (const float*)d_in[0];   // [16,2048,512] f32
    const float* cb   = (const float*)d_in[1];   // [4096,512]    f32
    const int N = in_sizes[0] / D_DIM;           // 32768 tokens

    // workspace layout: cb_hi (4MB) | cb_lo (4MB) | halfnorm (16KB)
    unsigned short* cbh = (unsigned short*)d_ws;
    unsigned short* cbl = cbh + (size_t)K_CODES * D_DIM;
    float* halfnorm = (float*)(cbl + (size_t)K_CODES * D_DIM);

    float* qout   = (float*)d_out;
    float* idxout = qout + (size_t)N * D_DIM;

    vq_pack_codebook<<<K_CODES, 256, 0, stream>>>(cb, cbh, cbl, halfnorm);

    const size_t smem = (size_t)2 * ROWS_WG * LDS_STRIDE * sizeof(unsigned short)
                      + ROWS_WG * sizeof(int);   // ~266.8 KB of the 320 KB WGP LDS
    (void)hipFuncSetAttribute((const void*)vq_main,
                              hipFuncAttributeMaxDynamicSharedMemorySize, (int)smem);
    vq_main<<<N / ROWS_WG, 256, smem, stream>>>(hsrc, cb, cbh, cbl, halfnorm,
                                                qout, idxout);
}
